// LSTMAutoEncoder_3040836845731
// MI455X (gfx1250) — matrix operand order repaired and checked by reading, not compiled, not run on hardware
//
#include <hip/hip_runtime.h>

// LSTM autoencoder for MI455X (gfx1250), wave32 + WMMA f16->f32.
// B=2048, S=512, F=32, H=64, L=3. One workgroup (128 thr = 4 waves) owns a
// 16-row batch tile and runs the whole recurrence with weights in LDS.
// x-tiles are prefetched with GLOBAL_LOAD_ASYNC_TO_LDS_B128 (double-buffered).

typedef _Float16 h16;
typedef __attribute__((ext_vector_type(16))) _Float16 v16h;
typedef __attribute__((ext_vector_type(8)))  float    v8f;

#define B_  2048
#define S_  512
#define F_  32
#define H_  64

struct Params {
  const float* in[27];   // ts, enc(Wih,Whh,bih,bhh)x3, dec(...)x3, W_out, b_out
  float* out;            // [B*S*F] output ++ [B*H] encoded
};

// ---- LDS layout (bytes), total 231552 < 320KB/WGP ----
#define OFF_WF    0u        // weight frags: 3 layers *16 nt *4 ks *32 lanes *16 h = 196608 B
#define OFF_WOUT  196608u   // W_out frags: 2 nt * 2 ks * 32 * 16 h              = 4096 B
#define OFF_XA    200704u   // layer-0 input A-frag tile (16x32 f16)             = 1024 B
#define OFF_HA    201728u   // h A-frags: 3 layers * 2 ktiles * 32 * 16 h        = 6144 B
#define OFF_GATES 207872u   // gate pre-acts: 4 gates * 16 m * 64 n f32          = 16384 B
#define OFF_BIAS  224256u   // bih+bhh per layer: 3 * 256 f32                    = 3072 B
#define OFF_BOUT  227328u   // b_out: 32 f32                                     = 128 B
#define OFF_XRAW  227456u   // raw x tiles, double buffered: 2 * 16*32 f32       = 4096 B
#define LDS_BYTES 231552u

__device__ __forceinline__ float sigm(float x) { return 1.0f / (1.0f + __expf(-x)); }

// Async DMA one 16x32-f32 x-tile (2048B) from ts into LDS buffer `buf`.
// 128 threads x 16B, tracked with ASYNCcnt (gfx1250 GLOBAL_LOAD_ASYNC_TO_LDS_B128).
__device__ __forceinline__ void async_copy_x(const float* ts, char* xraw,
                                             int b0, int t, int buf, int tid) {
  const int m  = tid >> 3;                 // batch row 0..15
  const int c4 = (tid & 7) * 4;            // float column base (16B chunk)
  const float* g = ts + ((size_t)(b0 + m) * S_ + t) * F_ + c4;
  const unsigned ldsoff =
      (unsigned)(uintptr_t)(xraw + buf * 2048 + m * 128 + (tid & 7) * 16);
  asm volatile("global_load_async_to_lds_b128 %0, %1, off"
               :: "v"(ldsoff), "v"((unsigned long long)(uintptr_t)g)
               : "memory");
}

// Load one phase's (encoder or decoder) 3-layer weights into LDS as B-fragments.
// B-frag layout (32x16 f16 tile): lane = n + 16*((k>>3)&1), elem = (k&7) + 8*(k>>4).
__device__ __forceinline__ void load_phase_weights(h16* Wf, float* biasS,
                                                   const Params& p, int base) {
  const int tid = threadIdx.x;
  for (int l = 0; l < 3; ++l) {
    const int din = (l == 0) ? F_ : H_;
    const int ksteps = (din + H_) / 32;           // 3 or 4
    const float* Wih = p.in[base + 4 * l + 0];    // [256, din]
    const float* Whh = p.in[base + 4 * l + 1];    // [256, 64]
    const float* bih = p.in[base + 4 * l + 2];
    const float* bhh = p.in[base + 4 * l + 3];
    for (int idx = tid; idx < 16 * 4 * 512; idx += 128) {
      const int elem = idx & 15;
      const int lane = (idx >> 4) & 31;
      const int ks   = (idx >> 9) & 3;
      const int nt   = idx >> 11;
      const int n  = nt * 16 + (lane & 15);                 // output column 0..255
      const int kk = 8 * (lane >> 4) + (elem & 7) + 16 * (elem >> 3);  // 0..31
      const int k  = ks * 32 + kk;
      float v = 0.0f;
      if (ks < ksteps) v = (k < din) ? Wih[n * din + k] : Whh[n * H_ + (k - din)];
      Wf[(size_t)l * (16 * 4 * 512) + idx] = (h16)v;
    }
    for (int idx = tid; idx < 256; idx += 128)
      biasS[l * 256 + idx] = bih[idx] + bhh[idx];
  }
}

// One LSTM layer for one time step: WMMA gates -> barrier -> elementwise cell.
// c8: this thread's 8 private cell-state values for layer l.
__device__ __forceinline__ void lstm_layer_step(
    const int l, const int lane, const int wv, const int em, const int enb,
    const h16* Wf, const h16* xA, h16* hA, float* gates, const float* biasS,
    float* c8, float* encDst) {
  const int ksteps = (l == 0) ? 3 : 4;   // K = din + 64 over 32-wide k-steps
  v8f acc[4];
  #pragma unroll
  for (int nt = 0; nt < 4; ++nt) acc[nt] = v8f{0.f,0.f,0.f,0.f,0.f,0.f,0.f,0.f};

  #pragma unroll
  for (int ks = 0; ks < 4; ++ks) {
    if (ks >= ksteps) break;
    // A-fragment source: [x | h_own] (l==0)  or  [h_below | h_own] (l>0)
    const h16* asrc;
    if (l == 0) asrc = (ks == 0) ? (xA + lane * 16)
                                 : (hA + (0 * 2 + (ks - 1)) * 512 + lane * 16);
    else        asrc = (ks < 2) ? (hA + ((l - 1) * 2 + ks) * 512 + lane * 16)
                                : (hA + (l * 2 + (ks - 2)) * 512 + lane * 16);
    const v16h a = *(const v16h*)asrc;
    #pragma unroll
    for (int nt = 0; nt < 4; ++nt) {
      const v16h b = *(const v16h*)(Wf + ((size_t)((l * 16 + wv * 4 + nt) * 4 + ks)) * 512
                                       + lane * 16);
      acc[nt] = __builtin_amdgcn_wmma_f32_16x16x32_f16(
          false, a, false, b, (short)0, acc[nt], false, false);
    }
  }

  // Scatter gate pre-activations (wave wv owns gate wv): C layout m = r + 8*(lane>>4)
  const int nl = lane & 15;
  const int mh = 8 * (lane >> 4);
  #pragma unroll
  for (int nt = 0; nt < 4; ++nt)
    #pragma unroll
    for (int r = 0; r < 8; ++r)
      gates[(wv * 16 + mh + r) * 64 + nt * 16 + nl] = acc[nt][r];

  __syncthreads();

  // Elementwise cell update: thread -> (m = em, n = enb..enb+7), c in registers.
  #pragma unroll
  for (int j = 0; j < 8; ++j) {
    const int n = enb + j;
    const float gi = gates[(0 * 16 + em) * 64 + n] + biasS[l * 256 + n];
    const float gf = gates[(1 * 16 + em) * 64 + n] + biasS[l * 256 + 64 + n];
    const float gg = gates[(2 * 16 + em) * 64 + n] + biasS[l * 256 + 128 + n];
    const float go = gates[(3 * 16 + em) * 64 + n] + biasS[l * 256 + 192 + n];
    const float c = sigm(gf) * c8[j] + sigm(gi) * tanhf(gg);
    c8[j] = c;
    const float h = sigm(go) * tanhf(c);
    // Store h into A-fragment layout of hA[l]:
    const int kk = n & 31, kt = n >> 5;
    const int la = em + 16 * ((kk >> 3) & 1);
    const int el = (kk & 7) + 8 * ((kk >> 4) & 1);
    hA[(l * 2 + kt) * 512 + la * 16 + el] = (h16)h;
    if (encDst) encDst[(size_t)em * H_ + n] = h;   // final-layer encoded output (f32)
  }
  __syncthreads();
}

__global__ __launch_bounds__(128) void lstm_ae_gfx1250(Params p) {
  extern __shared__ char smem[];
  h16*   Wf    = (h16*)(smem + OFF_WF);
  h16*   WoutF = (h16*)(smem + OFF_WOUT);
  h16*   xA    = (h16*)(smem + OFF_XA);
  h16*   hA    = (h16*)(smem + OFF_HA);
  float* gates = (float*)(smem + OFF_GATES);
  float* biasS = (float*)(smem + OFF_BIAS);
  float* boutS = (float*)(smem + OFF_BOUT);
  char*  xraw  = smem + OFF_XRAW;
  float* xrawf = (float*)xraw;

  const int tid  = threadIdx.x;
  const int lane = tid & 31;
  const int wv   = tid >> 5;          // 0..3, wave wv computes gate wv
  const int b0   = blockIdx.x * 16;   // batch tile base row
  const int em   = tid >> 3;          // elementwise row 0..15
  const int enb  = (tid & 7) * 8;     // elementwise col base

  const float* ts = p.in[0];

  // Kick off async DMA of x(0) immediately, before the weight staging work.
  async_copy_x(ts, xraw, b0, 0, 0, tid);

  // ---- one-time loads: W_out fragments + b_out ----
  for (int idx = tid; idx < 2048; idx += 128) {
    const int elem = idx & 15, la = (idx >> 4) & 31, ks = (idx >> 9) & 1, nt = idx >> 10;
    const int n  = nt * 16 + (la & 15);
    const int kk = 8 * (la >> 4) + (elem & 7) + 16 * (elem >> 3);
    WoutF[idx] = (h16)p.in[25][n * H_ + ks * 32 + kk];    // W_out[n][k], n<F, k<H
  }
  if (tid < F_) boutS[tid] = p.in[26][tid];

  load_phase_weights(Wf, biasS, p, /*encoder base=*/1);

  float creg[3][8];
  #pragma unroll
  for (int l = 0; l < 3; ++l)
    #pragma unroll
    for (int j = 0; j < 8; ++j) creg[l][j] = 0.0f;
  for (int idx = tid; idx < 3 * 2 * 512; idx += 128) hA[idx] = (h16)0.0f;
  __syncthreads();

  float* outp   = p.out;                                   // [B][S][F]
  float* outenc = p.out + (size_t)B_ * S_ * F_;            // [B][H]

  // ================= encoder: 512 steps =================
  for (int t = 0; t < S_; ++t) {
    // Prefetch x(t+1) into the other buffer, then wait for x(t) to land.
    if (t + 1 < S_) {
      async_copy_x(ts, xraw, b0, t + 1, (t + 1) & 1, tid);
      asm volatile("s_wait_asynccnt 0x1" ::: "memory");   // copy(t) complete (in-order)
    } else {
      asm volatile("s_wait_asynccnt 0x0" ::: "memory");
    }
    __syncthreads();   // publish async LDS writes from all waves

    // LDS->LDS: convert raw f32 x-tile into f16 A-fragment layout.
    const float* xb = xrawf + (t & 1) * 512;
    for (int idx = tid; idx < 512; idx += 128) {
      const int elem = idx & 15, la = idx >> 4;
      const int m  = la & 15;
      const int kk = (elem & 7) + 8 * (la >> 4) + 16 * (elem >> 3);
      xA[idx] = (h16)xb[m * 32 + kk];
    }
    __syncthreads();

    #pragma unroll
    for (int l = 0; l < 3; ++l) {
      float* ed = (l == 2 && t == S_ - 1) ? (outenc + (size_t)b0 * H_) : nullptr;
      lstm_layer_step(l, lane, wv, em, enb, Wf, xA, hA, gates, biasS, creg[l], ed);
    }
  }

  // ---- swap in decoder weights; reset c; keep h (= enc final states) ----
  __syncthreads();
  load_phase_weights(Wf, biasS, p, /*decoder base=*/13);
  #pragma unroll
  for (int l = 0; l < 3; ++l)
    #pragma unroll
    for (int j = 0; j < 8; ++j) creg[l][j] = 0.0f;
  __syncthreads();

  // ================= decoder: 512 steps (emit y, feed back) =================
  for (int i = 0; i < S_; ++i) {
    if (wv < 2) {   // waves 0,1: y = h_layer0 @ W_out^T + b_out  (16 x 32)
      v8f acc = v8f{0.f,0.f,0.f,0.f,0.f,0.f,0.f,0.f};
      #pragma unroll
      for (int ks = 0; ks < 2; ++ks) {
        const v16h a = *(const v16h*)(hA + (0 * 2 + ks) * 512 + lane * 16);
        const v16h b = *(const v16h*)(WoutF + ((wv * 2 + ks) * 32 + lane) * 16);
        acc = __builtin_amdgcn_wmma_f32_16x16x32_f16(
            false, a, false, b, (short)0, acc, false, false);
      }
      const int nl = lane & 15;
      const int n  = wv * 16 + nl;       // 0..31
      #pragma unroll
      for (int r = 0; r < 8; ++r) {
        const int m = r + 8 * (lane >> 4);
        const float y = acc[r] + boutS[n];
        // write-once stream: non-temporal store (TH=NT), keep L2 for weights/ts
        __builtin_nontemporal_store(
            y, &outp[((size_t)(b0 + m) * S_ + (S_ - 1 - i)) * F_ + n]);
        const int kk = n;                                            // 0..31
        const int la = m + 16 * ((kk >> 3) & 1);
        const int el = (kk & 7) + 8 * ((kk >> 4) & 1);
        xA[la * 16 + el] = (h16)y;                                   // feedback input
      }
    }
    __syncthreads();
    #pragma unroll
    for (int l = 0; l < 3; ++l)
      lstm_layer_step(l, lane, wv, em, enb, Wf, xA, hA, gates, biasS, creg[l], nullptr);
  }
}

extern "C" void kernel_launch(void* const* d_in, const int* in_sizes, int n_in,
                              void* d_out, int out_size, void* d_ws, size_t ws_size,
                              hipStream_t stream) {
  (void)in_sizes; (void)n_in; (void)out_size; (void)d_ws; (void)ws_size;
  Params p;
  for (int i = 0; i < 27; ++i) p.in[i] = (const float*)d_in[i];
  p.out = (float*)d_out;
  dim3 grid(B_ / 16), block(128);
  lstm_ae_gfx1250<<<grid, block, LDS_BYTES, stream>>>(p);
}
